// EnhancedHybridModel_49091476193526
// MI455X (gfx1250) — matrix operand order repaired and checked by reading, not compiled, not run on hardware
//
#include <hip/hip_runtime.h>
#include <hip/hip_fp16.h>

// ---------------------------------------------------------------------------
// EnhancedHybridModel for MI455X (gfx1250, wave32, WMMA).
// Memory-bound workload (~23 GFLOP, ~300MB f16 traffic): f16 activations,
// v_wmma_f32_16x16x32_f16 for conv2/conv3 (implicit GEMM), fc1, fc2;
// exact f32 VALU for conv1 (Cin=3), quantum statevector sim, and head.
// Conv GEMMs: templated shapes (all index math = shifts), boundary handling
// via clamped-address global loads + data cndmask (no flat/scratch traffic).
// Workspace requirement: ~62 MB.
// ---------------------------------------------------------------------------

typedef _Float16 f16;
typedef _Float16 v16h __attribute__((ext_vector_type(16)));
typedef float    v8f  __attribute__((ext_vector_type(8)));

constexpr float BN_INV = 0.9999950000374997f; // 1/sqrt(1+1e-5)

union FragU { v16h h; uint4 q[2]; };

#define WMMA_F16(a, b, c) \
  __builtin_amdgcn_wmma_f32_16x16x32_f16(false, (a), false, (b), (short)0, (c), false, false)

// A fragment (16x32 f16, MxK): lane = row M (lane&15); halves 0..7 hold
// K = kbase + hsel*8 .. +7, halves 8..15 hold K = kbase+16+hsel*8 .. +7
// (hsel = lane>>4). Two contiguous 16B loads.
__device__ __forceinline__ v16h loadA_rm(const f16* __restrict__ p, int K,
                                         int row, int kbase, int hsel) {
  FragU f;
  const f16* r = p + (size_t)row * K + kbase + hsel * 8;
  f.q[0] = *(const uint4*)(r);
  f.q[1] = *(const uint4*)(r + 16);
  return f.h;
}

// B fragment (32x16 f16, KxN), weights stored as Bt[N][K]: lane = column N
// (lane&15); halves 0..7 hold K = kbase + hsel*8 .. +7, halves 8..15 hold
// K = kbase+16+hsel*8 .. +7 (same as A). Two 16B loads.
__device__ __forceinline__ v16h loadB_rm(const f16* __restrict__ p, int K,
                                         int n, int kbase, int hsel) {
  FragU f;
  const f16* r = p + (size_t)n * K + kbase + hsel * 8;
  f.q[0] = *(const uint4*)(r);
  f.q[1] = *(const uint4*)(r + 16);
  return f.h;
}

// ---------------------------------------------------------------------------
// conv1 (3->32, 3x3 pad1) + BN + ReLU + 2x2 maxpool fused, exact f32 VALU.
// One thread per pooled output pixel; writes NHWC f16 [1024,16,16,32].
// ---------------------------------------------------------------------------
__global__ void conv1_pool_kernel(const float* __restrict__ x, const float* __restrict__ w,
                                  const float* __restrict__ bias, const float* __restrict__ gamma,
                                  const float* __restrict__ beta, f16* __restrict__ out) {
  __shared__ float sw[32 * 27];
  __shared__ float sscale[32], sshift[32];
  for (int i = threadIdx.x; i < 32 * 27; i += blockDim.x) sw[i] = w[i]; // [co][ci][ky][kx]
  if (threadIdx.x < 32) {
    float g = gamma[threadIdx.x];
    sscale[threadIdx.x] = g * BN_INV;
    sshift[threadIdx.x] = g * bias[threadIdx.x] * BN_INV + beta[threadIdx.x];
  }
  __syncthreads();
  int idx = blockIdx.x * blockDim.x + threadIdx.x;  // [0, 1024*16*16)
  int xo = idx & 15, yo = (idx >> 4) & 15, b = idx >> 8;
  float mx[32];
#pragma unroll
  for (int c = 0; c < 32; ++c) mx[c] = 0.0f;  // post-ReLU values are >= 0
#pragma unroll
  for (int py = 0; py < 2; ++py) {
#pragma unroll
    for (int px = 0; px < 2; ++px) {
      int yy = yo * 2 + py, xx = xo * 2 + px;
      float acc[32];
#pragma unroll
      for (int c = 0; c < 32; ++c) acc[c] = 0.0f;
      for (int ci = 0; ci < 3; ++ci) {
#pragma unroll
        for (int ky = 0; ky < 3; ++ky) {
          int iy = yy + ky - 1;
          if (iy < 0 || iy > 31) continue;
#pragma unroll
          for (int kx = 0; kx < 3; ++kx) {
            int ix = xx + kx - 1;
            if (ix < 0 || ix > 31) continue;
            float v = x[((b * 3 + ci) * 32 + iy) * 32 + ix];
            int t = ci * 9 + ky * 3 + kx;
#pragma unroll
            for (int co = 0; co < 32; ++co) acc[co] += v * sw[co * 27 + t];
          }
        }
      }
#pragma unroll
      for (int co = 0; co < 32; ++co) {
        float v = fmaxf(acc[co] * sscale[co] + sshift[co], 0.0f);
        mx[co] = fmaxf(mx[co], v);
      }
    }
  }
#pragma unroll
  for (int co = 0; co < 32; ++co) out[(size_t)idx * 32 + co] = (f16)mx[co];
}

// ---------------------------------------------------------------------------
// Implicit-GEMM conv (3x3 pad1) + BN + ReLU, NHWC f16 in/out, WMMA f16->f32.
// K loop = (patch position p, unrolled 9x) x (ci step 32). Boundary handling:
// clamp (iy,ix) to a valid pixel -> unconditional global_load_b128, then
// cndmask the loaded data against zero. No branches, no flat/scratch path.
// One wave = 32x32 C tile. blockDim = 128 (4 waves).
// ---------------------------------------------------------------------------
template <int BATCH, int H, int W, int Cin, int N>
__global__ void conv_gemm_kernel(const f16* __restrict__ A_in, const f16* __restrict__ Bt,
                                 const float* __restrict__ bias, const float* __restrict__ gamma,
                                 const float* __restrict__ beta, f16* __restrict__ out) {
  constexpr int K = 9 * Cin;
  constexpr int M = BATCH * H * W;
  constexpr int mWaves = M / 32, nWaves = N / 32;
  int lane = threadIdx.x & 31, hsel = lane >> 4, rowl = lane & 15;
  int gw = (blockIdx.x * blockDim.x + threadIdx.x) >> 5;
  if (gw >= mWaves * nWaves) return;   // wave-uniform: EXEC stays all-1s
  int mBase = (gw % mWaves) * 32;
  int nBase = (gw / mWaves) * 32;
  // Decompose the two A-rows this lane covers (W,H are powers of two -> shifts).
  int m0 = mBase + rowl, m1 = mBase + 16 + rowl;
  int x0 = m0 % W, y0 = (m0 / W) % H, b0 = m0 / (W * H);
  int x1 = m1 % W, y1 = (m1 / W) % H, b1 = m1 / (W * H);
  const f16* btRow0 = Bt + (size_t)(nBase + rowl) * K + hsel * 8;
  const f16* btRow1 = Bt + (size_t)(nBase + 16 + rowl) * K + hsel * 8;
  const uint4 z = {0u, 0u, 0u, 0u};
  v8f acc[2][2] = {};
#pragma unroll
  for (int p = 0; p < 9; ++p) {
    const int ky = p / 3 - 1, kx = p % 3 - 1;    // compile-time after unroll
    int iy0 = y0 + ky, ix0 = x0 + kx;
    int iy1 = y1 + ky, ix1 = x1 + kx;
    bool ok0 = (iy0 >= 0) & (iy0 < H) & (ix0 >= 0) & (ix0 < W);
    bool ok1 = (iy1 >= 0) & (iy1 < H) & (ix1 >= 0) & (ix1 < W);
    // clamped coordinates: address is always valid -> plain global loads
    int iyc0 = min(max(iy0, 0), H - 1), ixc0 = min(max(ix0, 0), W - 1);
    int iyc1 = min(max(iy1, 0), H - 1), ixc1 = min(max(ix1, 0), W - 1);
    const f16* p0 = A_in + ((size_t)((b0 * H + iyc0) * W + ixc0)) * Cin + hsel * 8;
    const f16* p1 = A_in + ((size_t)((b1 * H + iyc1) * W + ixc1)) * Cin + hsel * 8;
#pragma unroll
    for (int ci = 0; ci < Cin; ci += 32) {
      uint4 r00 = *(const uint4*)(p0 + ci);
      uint4 r01 = *(const uint4*)(p0 + ci + 16);
      uint4 r10 = *(const uint4*)(p1 + ci);
      uint4 r11 = *(const uint4*)(p1 + ci + 16);
      FragU a0, a1;
      a0.q[0] = ok0 ? r00 : z;
      a0.q[1] = ok0 ? r01 : z;
      a1.q[0] = ok1 ? r10 : z;
      a1.q[1] = ok1 ? r11 : z;
      const int k = p * Cin + Cin + ci - Cin;    // = p*Cin + ci
      FragU b0f, b1f;
      b0f.q[0] = *(const uint4*)(btRow0 + k);
      b0f.q[1] = *(const uint4*)(btRow0 + k + 16);
      b1f.q[0] = *(const uint4*)(btRow1 + k);
      b1f.q[1] = *(const uint4*)(btRow1 + k + 16);
      acc[0][0] = WMMA_F16(a0.h, b0f.h, acc[0][0]);
      acc[0][1] = WMMA_F16(a0.h, b1f.h, acc[0][1]);
      acc[1][0] = WMMA_F16(a1.h, b0f.h, acc[1][0]);
      acc[1][1] = WMMA_F16(a1.h, b1f.h, acc[1][1]);
    }
  }
#pragma unroll
  for (int tn = 0; tn < 2; ++tn) {
    int n = nBase + tn * 16 + rowl;
    float g = gamma[n];
    float scale = g * BN_INV;
    float shift = g * bias[n] * BN_INV + beta[n];
#pragma unroll
    for (int tm = 0; tm < 2; ++tm) {
#pragma unroll
      for (int r = 0; r < 8; ++r) {
        int m = mBase + tm * 16 + r + hsel * 8; // C layout: VGPR r -> M=r (+8 for hi lanes)
        float v = fmaxf(acc[tm][tn][r] * scale + shift, 0.0f);
        out[(size_t)m * N + n] = (f16)v;
      }
    }
  }
}

// ---------------------------------------------------------------------------
// Dense GEMM out = A[M,K] x Bt[N,K]^T, WMMA. EPI=0: +bias,ReLU -> f16.
// EPI=1: +bias -> f32. One wave = 32 x (16*TN) tile. Speculative prefetch one
// K-step ahead on the streamed B rows (global_prefetch path; OOB prefetches
// are dropped by hardware).
// ---------------------------------------------------------------------------
template <int TN, int EPI>
__global__ void dense_gemm_kernel(const f16* __restrict__ A, const f16* __restrict__ Bt,
                                  const float* __restrict__ bias, f16* __restrict__ out16,
                                  float* __restrict__ out32, int M, int N, int K) {
  int lane = threadIdx.x & 31, hsel = lane >> 4, rowl = lane & 15;
  int gw = (blockIdx.x * blockDim.x + threadIdx.x) >> 5;
  int mWaves = M / 32, nWaves = N / (16 * TN);
  if (gw >= mWaves * nWaves) return;
  int mBase = (gw % mWaves) * 32;
  int nBase = (gw / mWaves) * (16 * TN);
  v8f acc[2][TN] = {};
  for (int k = 0; k < K; k += 32) {
    v16h a0 = loadA_rm(A, K, mBase + rowl,      k, hsel);
    v16h a1 = loadA_rm(A, K, mBase + 16 + rowl, k, hsel);
#pragma unroll
    for (int tn = 0; tn < TN; ++tn) {
      __builtin_prefetch(Bt + (size_t)(nBase + tn * 16 + rowl) * K + k + 32, 0, 3);
      v16h b = loadB_rm(Bt, K, nBase + tn * 16 + rowl, k, hsel);
      acc[0][tn] = WMMA_F16(a0, b, acc[0][tn]);
      acc[1][tn] = WMMA_F16(a1, b, acc[1][tn]);
    }
  }
#pragma unroll
  for (int tn = 0; tn < TN; ++tn) {
    int n = nBase + tn * 16 + rowl;
    float bs = bias[n];
#pragma unroll
    for (int tm = 0; tm < 2; ++tm) {
#pragma unroll
      for (int r = 0; r < 8; ++r) {
        int m = mBase + tm * 16 + r + hsel * 8;
        float v = acc[tm][tn][r] + bs;
        if (EPI == 0) {
          out16[(size_t)m * N + n] = (f16)fmaxf(v, 0.0f);
        } else {
          out32[(size_t)m * N + n] = v;
        }
      }
    }
  }
}

// 2x2 maxpool, NHWC f16 (templated: index math is shifts).
template <int BATCH, int H, int W, int C>
__global__ void maxpool_nhwc(const f16* __restrict__ in, f16* __restrict__ out) {
  int idx = blockIdx.x * blockDim.x + threadIdx.x;
  constexpr int total = BATCH * (H / 2) * (W / 2) * C;
  if (idx >= total) return;
  int c = idx % C;
  int t = idx / C;
  int xo = t % (W / 2); t /= (W / 2);
  int yo = t % (H / 2);
  int b = t / (H / 2);
  const f16* p = in + ((size_t)((b * H + yo * 2) * W + xo * 2)) * C + c;
  float v0 = (float)p[0], v1 = (float)p[C];
  float v2 = (float)p[(size_t)W * C], v3 = (float)p[(size_t)W * C + C];
  out[idx] = (f16)fmaxf(fmaxf(v0, v1), fmaxf(v2, v3));
}

// 2x2 avgpool on NHWC [1024,8,8,128] + transpose to NCHW-flat [1024,2048] f16
// (matches the reference reshape: k = c*16 + y*4 + x).
__global__ void avgpool_t_kernel(const f16* __restrict__ in, f16* __restrict__ out) {
  int idx = blockIdx.x * blockDim.x + threadIdx.x;  // [0, 1024*2048)
  int k = idx & 2047, b = idx >> 11;
  int c = k >> 4, s = k & 15, y = s >> 2, x = s & 3;
  const f16* p = in + ((size_t)((b * 8 + y * 2) * 8 + x * 2)) * 128 + c;
  float v = (float)p[0] + (float)p[128] + (float)p[8 * 128] + (float)p[8 * 128 + 128];
  out[idx] = (f16)(v * 0.25f);
}

// Row softmax over 16 logits (f32).
__global__ void softmax16_kernel(const float* __restrict__ in, float* __restrict__ out) {
  int r = blockIdx.x * blockDim.x + threadIdx.x;
  if (r >= 1024) return;
  float v[16], mx = -1e30f;
#pragma unroll
  for (int j = 0; j < 16; ++j) { v[j] = in[r * 16 + j]; mx = fmaxf(mx, v[j]); }
  float s = 0.0f;
#pragma unroll
  for (int j = 0; j < 16; ++j) { v[j] = expf(v[j] - mx); s += v[j]; }
  float inv = 1.0f / s;
#pragma unroll
  for (int j = 0; j < 16; ++j) out[r * 16 + j] = v[j] * inv;
}

// ---------------------------------------------------------------------------
// 4-qubit statevector sim: amplitude-encode softmax row, 2x (RX kron + CNOT
// ring), Pauli-Z expvals. U-kron matrices built cooperatively in LDS; the
// CNOT ring is a fixed basis permutation baked in below.
// ---------------------------------------------------------------------------
__global__ void quantum_kernel(const float* __restrict__ f, const float* __restrict__ qw,
                               float* __restrict__ qz) {
  __shared__ float Ur[2][16][16], Ui[2][16][16];
  __shared__ float sc[2][4], ss[2][4];
  if (threadIdx.x < 8) {
    int l = threadIdx.x >> 2, q = threadIdx.x & 3;
    float th = qw[l * 4 + q] * 0.5f;
    sc[l][q] = cosf(th);
    ss[l][q] = sinf(th);
  }
  __syncthreads();
  for (int e = threadIdx.x; e < 512; e += blockDim.x) {
    int l = e >> 8, m = (e >> 4) & 15, j = e & 15;
    float re = 1.0f, im = 0.0f;
#pragma unroll
    for (int q = 0; q < 4; ++q) {
      int bm = (m >> (3 - q)) & 1, bj = (j >> (3 - q)) & 1;
      if (bm == bj) { re *= sc[l][q]; im *= sc[l][q]; }
      else { float s = ss[l][q]; float nr = im * s, ni = -re * s; re = nr; im = ni; } // *(-i s)
    }
    Ur[l][m][j] = re; Ui[l][m][j] = im;
  }
  __syncthreads();
  int row = blockIdx.x * blockDim.x + threadIdx.x;  // 0..1023
  float pr[16], pim[16], nrm = 0.0f;
#pragma unroll
  for (int j = 0; j < 16; ++j) { float v = fabsf(f[row * 16 + j]); pr[j] = v; nrm += v * v; }
  float inv = rsqrtf(nrm);
#pragma unroll
  for (int j = 0; j < 16; ++j) { pr[j] *= inv; pim[j] = 0.0f; }
  // psi'[PERM[m]] = (Ukron @ psi)[m]  (PERM = CNOT ring 0->1->2->3->0, wire0=MSB)
  constexpr int PERM[16] = {0, 9, 11, 2, 15, 6, 4, 13, 7, 14, 12, 5, 8, 1, 3, 10};
#pragma unroll
  for (int l = 0; l < 2; ++l) {
    float tr[16], ti[16];
#pragma unroll
    for (int m = 0; m < 16; ++m) {
      float ar = 0.0f, ai = 0.0f;
#pragma unroll
      for (int j = 0; j < 16; ++j) {
        float ur = Ur[l][m][j], ui = Ui[l][m][j];
        ar += ur * pr[j] - ui * pim[j];
        ai += ur * pim[j] + ui * pr[j];
      }
      tr[m] = ar; ti[m] = ai;
    }
#pragma unroll
    for (int m = 0; m < 16; ++m) { pr[PERM[m]] = tr[m]; pim[PERM[m]] = ti[m]; }
  }
#pragma unroll
  for (int w = 0; w < 4; ++w) {
    float e = 0.0f;
#pragma unroll
    for (int b = 0; b < 16; ++b) {
      float p = pr[b] * pr[b] + pim[b] * pim[b];
      e += (((b >> (3 - w)) & 1) ? -p : p);
    }
    qz[row * 4 + w] = e;
  }
}

// head: hidden = relu(bn(q @ h1_w^T + h1_b))  -> f32 [1024,128]
__global__ void head_hidden_kernel(const float* __restrict__ qz, const float* __restrict__ h1w,
                                   const float* __restrict__ h1b, const float* __restrict__ g,
                                   const float* __restrict__ bt, float* __restrict__ hid) {
  int idx = blockIdx.x * blockDim.x + threadIdx.x;  // [0, 1024*128)
  int b = idx >> 7, j = idx & 127;
  float v = h1b[j];
#pragma unroll
  for (int k = 0; k < 4; ++k) v += qz[b * 4 + k] * h1w[j * 4 + k];
  v = g[j] * v * BN_INV + bt[j];
  hid[idx] = fmaxf(v, 0.0f);
}

// out = hidden @ h2_w^T + h2_b -> f32 [1024,100]
__global__ void head_out_kernel(const float* __restrict__ hid, const float* __restrict__ h2w,
                                const float* __restrict__ h2b, float* __restrict__ out) {
  int idx = blockIdx.x * blockDim.x + threadIdx.x;  // [0, 1024*100)
  int b = idx / 100, o = idx % 100;
  float a = h2b[o];
  for (int j = 0; j < 128; ++j) a += hid[b * 128 + j] * h2w[o * 128 + j];
  out[idx] = a;
}

// Weight prep: conv w[Cout,Cin,3,3] f32 -> Bt[co][ (ky*3+kx)*Cin + ci ] f16.
__global__ void prep_conv_w_kernel(const float* __restrict__ w, f16* __restrict__ bt,
                                   int Cout, int Cin) {
  int idx = blockIdx.x * blockDim.x + threadIdx.x;
  int K = 9 * Cin;
  if (idx >= Cout * K) return;
  int co = idx / K, rem = idx % K;
  int p = rem / Cin, ci = rem % Cin;
  int ky = p / 3, kx = p % 3;
  bt[idx] = (f16)w[((co * Cin + ci) * 3 + ky) * 3 + kx];
}

__global__ void f32_to_f16_kernel(const float* __restrict__ a, f16* __restrict__ b, int n) {
  int idx = blockIdx.x * blockDim.x + threadIdx.x;
  if (idx < n) b[idx] = (f16)a[idx];
}

// ---------------------------------------------------------------------------
// Workspace layout (~62 MB total)
// ---------------------------------------------------------------------------
constexpr size_t ALN(size_t x) { return (x + 255) & ~(size_t)255; }
constexpr size_t OFF_W2  = 0;                                               // conv2 Bt f16 [64,288]
constexpr size_t OFF_W3  = OFF_W2  + ALN((size_t)64 * 288 * 2);             // conv3 Bt f16 [128,576]
constexpr size_t OFF_WF1 = OFF_W3  + ALN((size_t)128 * 576 * 2);            // fc1 Bt f16 [512,2048]
constexpr size_t OFF_WF2 = OFF_WF1 + ALN((size_t)512 * 2048 * 2);           // fc2 Bt f16 [16,512]
constexpr size_t OFF_R1  = OFF_WF2 + ALN((size_t)16 * 512 * 2);             // 16MB: act1p / act3 / fc1out
constexpr size_t OFF_R2  = OFF_R1  + ALN((size_t)1024 * 16 * 16 * 32 * 2);  // 32MB: act2
constexpr size_t OFF_R3  = OFF_R2  + ALN((size_t)1024 * 16 * 16 * 64 * 2);  // 8MB: act2p / act3p
constexpr size_t OFF_LOG = OFF_R3  + ALN((size_t)1024 * 8 * 8 * 64 * 2);    // fc2 logits f32
constexpr size_t OFF_F   = OFF_LOG + ALN((size_t)1024 * 16 * 4);            // softmax f32
constexpr size_t OFF_QZ  = OFF_F   + ALN((size_t)1024 * 16 * 4);            // Z expvals f32
constexpr size_t OFF_HID = OFF_QZ  + ALN((size_t)1024 * 4 * 4);             // head hidden f32

extern "C" void kernel_launch(void* const* d_in, const int* in_sizes, int n_in,
                              void* d_out, int out_size, void* d_ws, size_t ws_size,
                              hipStream_t stream) {
  (void)in_sizes; (void)n_in; (void)out_size; (void)ws_size;
  const float* x       = (const float*)d_in[0];
  const float* conv1_w = (const float*)d_in[1];
  const float* conv1_b = (const float*)d_in[2];
  const float* bn1_g   = (const float*)d_in[3];
  const float* bn1_b   = (const float*)d_in[4];
  const float* conv2_w = (const float*)d_in[5];
  const float* conv2_b = (const float*)d_in[6];
  const float* bn2_g   = (const float*)d_in[7];
  const float* bn2_b   = (const float*)d_in[8];
  const float* conv3_w = (const float*)d_in[9];
  const float* conv3_b = (const float*)d_in[10];
  const float* bn3_g   = (const float*)d_in[11];
  const float* bn3_b   = (const float*)d_in[12];
  const float* fr1_w   = (const float*)d_in[13];
  const float* fr1_b   = (const float*)d_in[14];
  const float* fr2_w   = (const float*)d_in[15];
  const float* fr2_b   = (const float*)d_in[16];
  const float* q_w     = (const float*)d_in[17];
  const float* h1_w    = (const float*)d_in[18];
  const float* h1_b    = (const float*)d_in[19];
  const float* bnh_g   = (const float*)d_in[20];
  const float* bnh_b   = (const float*)d_in[21];
  const float* h2_w    = (const float*)d_in[22];
  const float* h2_b    = (const float*)d_in[23];

  char* ws = (char*)d_ws;
  f16*   W2   = (f16*)(ws + OFF_W2);
  f16*   W3   = (f16*)(ws + OFF_W3);
  f16*   WF1  = (f16*)(ws + OFF_WF1);
  f16*   WF2  = (f16*)(ws + OFF_WF2);
  f16*   R1   = (f16*)(ws + OFF_R1);
  f16*   R2   = (f16*)(ws + OFF_R2);
  f16*   R3   = (f16*)(ws + OFF_R3);
  float* LOG  = (float*)(ws + OFF_LOG);
  float* FBUF = (float*)(ws + OFF_F);
  float* QZ   = (float*)(ws + OFF_QZ);
  float* HID  = (float*)(ws + OFF_HID);

  // --- weight preparation (f32 -> f16, conv weights to [Cout, K] im2col order)
  prep_conv_w_kernel<<<(64 * 288 + 255) / 256, 256, 0, stream>>>(conv2_w, W2, 64, 32);
  prep_conv_w_kernel<<<(128 * 576 + 255) / 256, 256, 0, stream>>>(conv3_w, W3, 128, 64);
  f32_to_f16_kernel<<<(512 * 2048) / 256, 256, 0, stream>>>(fr1_w, WF1, 512 * 2048);
  f32_to_f16_kernel<<<(16 * 512 + 255) / 256, 256, 0, stream>>>(fr2_w, WF2, 16 * 512);

  // --- conv1 + BN + ReLU + maxpool (fused, f32 VALU) -> R1 NHWC [1024,16,16,32]
  conv1_pool_kernel<<<1024, 256, 0, stream>>>(x, conv1_w, conv1_b, bn1_g, bn1_b, R1);

  // --- conv2 (WMMA implicit GEMM, M=262144 K=288 N=64) -> R2 [1024,16,16,64]
  conv_gemm_kernel<1024, 16, 16, 32, 64><<<4096, 128, 0, stream>>>(
      R1, W2, conv2_b, bn2_g, bn2_b, R2);
  // --- maxpool -> R3 [1024,8,8,64]
  maxpool_nhwc<1024, 16, 16, 64><<<(1024 * 8 * 8 * 64) / 256, 256, 0, stream>>>(R2, R3);

  // --- conv3 (WMMA, M=65536 K=576 N=128) -> R1 [1024,8,8,128]
  conv_gemm_kernel<1024, 8, 8, 64, 128><<<2048, 128, 0, stream>>>(
      R3, W3, conv3_b, bn3_g, bn3_b, R1);
  // --- avgpool + NCHW transpose -> R3 [1024,2048] f16 (matches reference reshape)
  avgpool_t_kernel<<<(1024 * 2048) / 256, 256, 0, stream>>>(R1, R3);

  // --- fc1 (WMMA, M=1024 K=2048 N=512, +bias +ReLU) -> R1 f16 [1024,512]
  dense_gemm_kernel<2, 0><<<128, 128, 0, stream>>>(R3, WF1, fr1_b, R1, nullptr,
                                                   1024, 512, 2048);
  // --- fc2 (WMMA, M=1024 K=512 N=16, +bias) -> LOG f32 [1024,16]
  dense_gemm_kernel<1, 1><<<8, 128, 0, stream>>>(R1, WF2, fr2_b, nullptr, LOG,
                                                 1024, 16, 512);
  // --- softmax over 16 -> FBUF
  softmax16_kernel<<<4, 256, 0, stream>>>(LOG, FBUF);

  // --- quantum layer -> QZ [1024,4]
  quantum_kernel<<<4, 256, 0, stream>>>(FBUF, q_w, QZ);

  // --- head
  head_hidden_kernel<<<(1024 * 128) / 256, 256, 0, stream>>>(QZ, h1_w, h1_b, bnh_g, bnh_b, HID);
  head_out_kernel<<<(1024 * 100) / 256, 256, 0, stream>>>(HID, h2_w, h2_b, (float*)d_out);
}
